// LinearAttention_15771119910915
// MI455X (gfx1250) — matrix operand order repaired and checked by reading, not compiled, not run on hardware
//
#include <hip/hip_runtime.h>
#include <hip/hip_bf16.h>

typedef __attribute__((ext_vector_type(16))) __bf16 v16bf;
typedef __attribute__((ext_vector_type(8)))  __bf16 v8bf;
typedef __attribute__((ext_vector_type(8)))  float  v8f;
typedef __attribute__((ext_vector_type(4)))  int    v4i;

#define cB  4
#define cT  4096
#define cD  1024
#define cH  16
#define cDH 64
#define cBT (cB * cT)
#define LPITCH 40   // LDS row pitch (bf16 halves): 80B, 16B-aligned, 20-bank stride
#define QPITCH 72   // out-kernel q tile pitch: 144B, 16B-aligned, 36-bank stride

// ---------------------------------------------------------------------------
// CDNA5 async global->LDS copy (16B per lane), ASYNCcnt-tracked
// ---------------------------------------------------------------------------
__device__ __forceinline__ void async_copy16(const void* g, void* l) {
#if __has_builtin(__builtin_amdgcn_global_load_async_to_lds_b128)
    __builtin_amdgcn_global_load_async_to_lds_b128(
        (__attribute__((address_space(1))) v4i*)g,
        (__attribute__((address_space(3))) v4i*)l, 0, 0);
#else
    unsigned lds_off = (unsigned)(size_t)(__attribute__((address_space(3))) void*)l;
    asm volatile("global_load_async_to_lds_b128 %0, %1, off"
                 :: "v"(lds_off), "v"((unsigned long long)(size_t)g)
                 : "memory");
#endif
}

template <int N>
__device__ __forceinline__ void wait_async() {
#if __has_builtin(__builtin_amdgcn_s_wait_asynccnt)
    __builtin_amdgcn_s_wait_asynccnt(N);
#else
    asm volatile("s_wait_asynccnt %0" :: "n"(N) : "memory");
#endif
}

// ---------------------------------------------------------------------------
// fp32 -> bf16 conversion (8 elements / thread), for x
// ---------------------------------------------------------------------------
__global__ __launch_bounds__(256) void la_cvt_bf16(const float* __restrict__ s,
                                                   __bf16* __restrict__ d) {
    size_t i = ((size_t)blockIdx.x * 256 + threadIdx.x) * 8;
    float4 a = *(const float4*)(s + i);
    float4 b = *(const float4*)(s + i + 4);
    v8bf o;
    o[0] = (__bf16)a.x; o[1] = (__bf16)a.y; o[2] = (__bf16)a.z; o[3] = (__bf16)a.w;
    o[4] = (__bf16)b.x; o[5] = (__bf16)b.y; o[6] = (__bf16)b.z; o[7] = (__bf16)b.w;
    *(v8bf*)(d + i) = o;
}

// ---------------------------------------------------------------------------
// fp32 W[k][n] -> bf16 Wt[n][k] (convert + transpose), 64x64 LDS tiles
// ---------------------------------------------------------------------------
__global__ __launch_bounds__(256) void la_cvt_w_t(const float* __restrict__ W,
                                                  __bf16* __restrict__ Wt) {
    __shared__ float tile[64][65];
    const int bx = blockIdx.x * 64;   // k base
    const int by = blockIdx.y * 64;   // n base
    #pragma unroll
    for (int it = 0; it < 16; ++it) {
        int idx = threadIdx.x + it * 256;
        int r = idx >> 6, c = idx & 63;
        tile[r][c] = W[(size_t)(bx + r) * cD + by + c];
    }
    __syncthreads();
    #pragma unroll
    for (int it = 0; it < 16; ++it) {
        int idx = threadIdx.x + it * 256;
        int r = idx >> 6, c = idx & 63;      // r: n offset, c: k offset
        Wt[(size_t)(by + r) * cD + bx + c] = (__bf16)tile[c][r];
    }
}

// ---------------------------------------------------------------------------
// Fused projection GEMM: [BT x D] @ [D x 3D] (+bias, elu+1 on q/k)
// 256 thr (8 waves), tile 128x128, k-step 32, double-buffered async staging.
// Weights pre-transposed (Wt[n][k]) so both LDS tiles stage contiguously.
// k and v are stored transposed per head for the downstream kv GEMM.
// ---------------------------------------------------------------------------
__global__ __launch_bounds__(256) void la_proj(const __bf16* __restrict__ xb,
                                               const __bf16* __restrict__ Wt,
                                               const float* __restrict__ pbq,
                                               const float* __restrict__ pbk,
                                               const float* __restrict__ pbv,
                                               __bf16* __restrict__ qb,
                                               __bf16* __restrict__ kbt,   // [B*H][DH][T]
                                               __bf16* __restrict__ vbt) { // [B*H][DH][T]
    __shared__ __bf16 As[2][128 * LPITCH];   // A tile [m][k]
    __shared__ __bf16 Bst[2][128 * LPITCH];  // B tile [n][k]

    const int tid = threadIdx.x;
    const int m0  = blockIdx.x * 128;          // row tile in [0, BT)
    const int n0g = blockIdx.y * 128;          // col tile in [0, 3D)
    const int w   = n0g / cD;                  // 0=q, 1=k, 2=v (uniform per block)
    const int c0  = n0g % cD;
    const __bf16* Wsl = Wt + (size_t)w * cD * cD;

    const int sr = tid >> 2;                   // staging row 0..63 (x2 chunks)
    const int sc = (tid & 3) * 8;              // staging col offset (halves)

    const int wid = tid >> 5, lane = tid & 31;
    const int wm = wid & 3, wn = wid >> 2;     // 4 (M) x 2 (N) wave grid
    const int fm  = lane & 15;
    const int kb8 = (lane >> 4) * 8;           // A-frag K base
    const int kh  = (lane >> 4) * 16;          // unused; B-frag also uses kb8

    v8f acc[2][4] = {};

    // stage tile kk into buffer buf (4 async ops / thread: 2 A + 2 B)
    auto stage = [&](int buf, int kk) {
        #pragma unroll
        for (int it = 0; it < 2; ++it) {
            int r = sr + it * 64;
            async_copy16(xb  + (size_t)(m0 + r) * cD + kk + sc, &As[buf][r * LPITCH + sc]);
            async_copy16(Wsl + (size_t)(c0 + r) * cD + kk + sc, &Bst[buf][r * LPITCH + sc]);
        }
    };

    stage(0, 0);
    int buf = 0;
    for (int kk = 0; kk < cD; kk += 32) {
        // branch-free pipeline: last iteration redundantly re-stages tile 0
        // into the dead buffer; S_ENDPGM's implicit wait-idle covers it.
        stage(buf ^ 1, (kk + 32) & (cD - 1));
        wait_async<4>();                 // current tile landed; next 4 in flight
        __syncthreads();

        v16bf afr[2], bfr[4];
        #pragma unroll
        for (int i = 0; i < 2; ++i) {
            const __bf16* p = &As[buf][(wm * 32 + i * 16 + fm) * LPITCH + kb8];
            *((v8bf*)&afr[i])     = *(const v8bf*)(p);        // K kb8..kb8+7
            *((v8bf*)&afr[i] + 1) = *(const v8bf*)(p + 16);   // K kb8+16..kb8+23
        }
        #pragma unroll
        for (int j = 0; j < 4; ++j) {
            const __bf16* p = &Bst[buf][(wn * 64 + j * 16 + fm) * LPITCH + kb8];
            *((v8bf*)&bfr[j])     = *(const v8bf*)(p);        // K kb8..kb8+7
            *((v8bf*)&bfr[j] + 1) = *(const v8bf*)(p + 16);   // K kb8+16..kb8+23
        }
        #pragma unroll
        for (int i = 0; i < 2; ++i)
            #pragma unroll
            for (int j = 0; j < 4; ++j)
                acc[i][j] = __builtin_amdgcn_wmma_f32_16x16x32_bf16(
                    false, afr[i], false, bfr[j], (short)0, acc[i][j], false, false);
        __syncthreads();
        buf ^= 1;
    }

    // epilogue: bias, elu+1 (q,k); k and v stored transposed per head
    const float* bias_ptr = (w == 0) ? pbq : ((w == 1) ? pbk : pbv);
    #pragma unroll
    for (int i = 0; i < 2; ++i) {
        #pragma unroll
        for (int j = 0; j < 4; ++j) {
            int colw = c0 + wn * 64 + j * 16 + (lane & 15);
            float bias = bias_ptr[colw];
            #pragma unroll
            for (int r = 0; r < 8; ++r) {
                int row = m0 + wm * 32 + i * 16 + r + ((lane >> 4) << 3);
                float val = acc[i][j][r] + bias;
                if (w < 2) val = (val > 0.f) ? (val + 1.f) : __expf(val);  // elu(x)+1
                __bf16 bvv = (__bf16)val;
                if (w == 0) {
                    qb[(size_t)row * cD + colw] = bvv;
                } else {
                    int b = row >> 12, t = row & (cT - 1);
                    int h = colw >> 6, dd = colw & 63;
                    size_t idx = (((size_t)(b * cH + h) * cDH + dd) << 12) + t;
                    if (w == 1) kbt[idx] = bvv; else vbt[idx] = bvv;
                }
            }
        }
    }
}

// ---------------------------------------------------------------------------
// kv = k^T @ v per (b,h): M=N=64, K=T=4096; also ksum[d] = sum_t k[t][d]
// Both operands stored [dim][t] -> contiguous async staging, no transposes.
// kv written transposed [bh][e][d] for the out-GEMM B-fragments.
// ---------------------------------------------------------------------------
__global__ __launch_bounds__(256) void la_kv(const __bf16* __restrict__ kbt,
                                             const __bf16* __restrict__ vbt,
                                             float* __restrict__ ksum,
                                             __bf16* __restrict__ kvt) {
    __shared__ __bf16 As[2][64 * LPITCH];
    __shared__ __bf16 Bst[2][64 * LPITCH];
    __shared__ float  red[256];

    const int tid = threadIdx.x;
    const int bh  = blockIdx.x;                 // 0..63
    const __bf16* kT = kbt + ((size_t)bh << 18);   // [64][4096]
    const __bf16* vT = vbt + ((size_t)bh << 18);   // [64][4096]

    // ---- ksum: row sums of kT ----
    {
        const int d = tid >> 2, q = tid & 3;
        const __bf16* p = kT + (size_t)d * cT + q * 1024;
        float s = 0.f;
        for (int i = 0; i < 1024; i += 8) {
            v8bf x = *(const v8bf*)(p + i);
            #pragma unroll
            for (int j = 0; j < 8; ++j) s += (float)x[j];
        }
        red[tid] = s;
    }
    __syncthreads();
    if ((tid & 3) == 0)
        ksum[bh * cDH + (tid >> 2)] = red[tid] + red[tid + 1] + red[tid + 2] + red[tid + 3];
    __syncthreads();

    const int sr = tid >> 2, sc = (tid & 3) * 8;   // 64 rows x 4 chunks
    const int wid = tid >> 5, lane = tid & 31;
    const int wm = wid & 3, wn = wid >> 2;
    const int fm  = lane & 15;
    const int kb8 = (lane >> 4) * 8;
    const int kh  = (lane >> 4) * 16;
    v8f acc[2] = {};

    auto stage = [&](int bf, int kk) {   // 2 async ops / thread
        async_copy16(kT + (size_t)sr * cT + kk + sc, &As[bf][sr * LPITCH + sc]);
        async_copy16(vT + (size_t)sr * cT + kk + sc, &Bst[bf][sr * LPITCH + sc]);
    };

    stage(0, 0);
    int buf = 0;
    for (int kk = 0; kk < cT; kk += 32) {
        stage(buf ^ 1, (kk + 32) & (cT - 1));   // branch-free (see la_proj)
        wait_async<2>();
        __syncthreads();

        v16bf afr;
        {
            const __bf16* p = &As[buf][(wm * 16 + fm) * LPITCH + kb8];
            *((v8bf*)&afr)     = *(const v8bf*)(p);
            *((v8bf*)&afr + 1) = *(const v8bf*)(p + 16);
        }
        #pragma unroll
        for (int j = 0; j < 2; ++j) {
            const __bf16* p = &Bst[buf][(wn * 32 + j * 16 + fm) * LPITCH + kb8];
            v16bf bfr;
            *((v8bf*)&bfr)     = *(const v8bf*)(p);
            *((v8bf*)&bfr + 1) = *(const v8bf*)(p + 16);
            acc[j] = __builtin_amdgcn_wmma_f32_16x16x32_bf16(
                false, afr, false, bfr, (short)0, acc[j], false, false);
        }
        __syncthreads();
        buf ^= 1;
    }

    #pragma unroll
    for (int j = 0; j < 2; ++j) {
        int n = wn * 32 + j * 16 + fm;
        #pragma unroll
        for (int r = 0; r < 8; ++r) {
            int m = wm * 16 + r + ((lane >> 4) << 3);
            kvt[((size_t)bh << 12) + n * cDH + m] = (__bf16)acc[j][r];   // [e][d]
        }
    }
}

// ---------------------------------------------------------------------------
// out = (q @ kv) * Dinv, Dinv = 1/(q . (ksum+1e-6)); fp32 output
// grid (B*H, T/128); q tile async-staged to LDS; kv fragments from global (hot)
// ---------------------------------------------------------------------------
__global__ __launch_bounds__(256) void la_out(const __bf16* __restrict__ qb,
                                              const __bf16* __restrict__ kvt,
                                              const float* __restrict__ ksum,
                                              float* __restrict__ outp) {
    __shared__ __bf16 qs[128 * QPITCH];
    __shared__ float s_ks[cDH];
    __shared__ float s_dinv[128];

    const int tid = threadIdx.x;
    const int bh = blockIdx.x;
    const int b = bh >> 4, h = bh & 15;
    const int t0 = blockIdx.y * 128;
    const __bf16* qbase = qb + (size_t)(b * cT + t0) * cD + h * cDH;

    // stage q tile: 128 rows x 64 halves, 4 async chunks / thread
    #pragma unroll
    for (int it = 0; it < 4; ++it) {
        int chunk = tid + it * 256;          // 0..1023
        int r = chunk >> 3, cc = (chunk & 7) * 8;
        async_copy16(qbase + (size_t)r * cD + cc, &qs[r * QPITCH + cc]);
    }
    if (tid < cDH) s_ks[tid] = ksum[bh * cDH + tid] + 1e-6f;
    wait_async<0>();
    __syncthreads();

    if (tid < 128) {
        float s = 0.f;
        #pragma unroll
        for (int i = 0; i < cDH; i += 8) {
            v8bf x = *(const v8bf*)(&qs[tid * QPITCH + i]);
            #pragma unroll
            for (int j = 0; j < 8; ++j) s += (float)x[j] * s_ks[i + j];
        }
        s_dinv[tid] = 1.0f / s;
    }
    __syncthreads();

    const int wid = tid >> 5, lane = tid & 31;
    const int fm  = lane & 15;
    const int kb8 = (lane >> 4) * 8;
    const int kh  = (lane >> 4) * 16;
    v8f acc[4] = {};
    const __bf16* kvs = kvt + ((size_t)bh << 12);

    #pragma unroll
    for (int ks = 0; ks < 2; ++ks) {     // K = 64 in two 32-steps
        v16bf afr;
        const __bf16* p = &qs[(wid * 16 + fm) * QPITCH + ks * 32 + kb8];
        *((v8bf*)&afr)     = *(const v8bf*)(p);
        *((v8bf*)&afr + 1) = *(const v8bf*)(p + 16);
        #pragma unroll
        for (int j = 0; j < 4; ++j) {
            const __bf16* bp = kvs + (j * 16 + fm) * cDH + ks * 32 + kb8;  // [e][d]
            v16bf bfr;
            *((v8bf*)&bfr)     = *(const v8bf*)(bp);
            *((v8bf*)&bfr + 1) = *(const v8bf*)(bp + 16);
            acc[j] = __builtin_amdgcn_wmma_f32_16x16x32_bf16(
                false, afr, false, bfr, (short)0, acc[j], false, false);
        }
    }

    #pragma unroll
    for (int j = 0; j < 4; ++j) {
        #pragma unroll
        for (int r = 0; r < 8; ++r) {
            int rl = wid * 16 + r + ((lane >> 4) << 3);     // local row 0..127
            float val = acc[j][r] * s_dinv[rl];
            outp[(size_t)(b * cT + t0 + rl) * cD + h * cDH + j * 16 + (lane & 15)] = val;
        }
    }
}

// ---------------------------------------------------------------------------
extern "C" void kernel_launch(void* const* d_in, const int* in_sizes, int n_in,
                              void* d_out, int out_size, void* d_ws, size_t ws_size,
                              hipStream_t stream) {
    (void)in_sizes; (void)n_in; (void)out_size; (void)ws_size;
    const float* x  = (const float*)d_in[0];
    const float* Wq = (const float*)d_in[1];
    const float* bq = (const float*)d_in[2];
    const float* Wk = (const float*)d_in[3];
    const float* bk = (const float*)d_in[4];
    const float* Wv = (const float*)d_in[5];
    const float* bv = (const float*)d_in[6];
    float* outp = (float*)d_out;

    char* p = (char*)d_ws;
    __bf16* xb   = (__bf16*)p; p += (size_t)cBT * cD * 2;
    __bf16* Wt   = (__bf16*)p; p += (size_t)3 * cD * cD * 2;
    __bf16* qb   = (__bf16*)p; p += (size_t)cBT * cD * 2;
    __bf16* kbt  = (__bf16*)p; p += (size_t)cBT * cD * 2;
    __bf16* vbt  = (__bf16*)p; p += (size_t)cBT * cD * 2;
    __bf16* kvt  = (__bf16*)p; p += (size_t)cB * cH * cDH * cDH * 2;
    float*  ksum = (float*)p;  p += (size_t)cB * cH * cDH * 4;

    la_cvt_bf16<<<(cBT * cD) / 2048, 256, 0, stream>>>(x, xb);
    la_cvt_w_t<<<dim3(16, 16), 256, 0, stream>>>(Wq, Wt);
    la_cvt_w_t<<<dim3(16, 16), 256, 0, stream>>>(Wk, Wt + (size_t)cD * cD);
    la_cvt_w_t<<<dim3(16, 16), 256, 0, stream>>>(Wv, Wt + (size_t)2 * cD * cD);

    la_proj<<<dim3(cBT / 128, (3 * cD) / 128), 256, 0, stream>>>(
        xb, Wt, bq, bk, bv, qb, kbt, vbt);

    la_kv<<<cB * cH, 256, 0, stream>>>(kbt, vbt, ksum, kvt);

    la_out<<<dim3(cB * cH, cT / 128), 256, 0, stream>>>(qb, kvt, ksum, outp);
}
